// Attention_77154792505463
// MI455X (gfx1250) — matrix operand order repaired and checked by reading, not compiled, not run on hardware
//
#include <hip/hip_runtime.h>
#include <hip/hip_bf16.h>

// ---------------------------------------------------------------------------
// Attention for B=4,H=16,S=2048,D=64 (fp32), returning (result, softmax scores)
// One workgroup (8 waves / 256 threads) per (b,h, 16-query-row block).
// Score stripe (16 x 2048 fp32 = 128 KB) staged in CDNA5's large LDS so the
// S*S score matrix is written to HBM exactly once (memory-roofline bound).
// QK^T uses v_wmma_f32_16x16x32_bf16 with 3-term bf16 error compensation
// (hi*hi + hi*lo + lo*hi); P*V uses 2-term (P compensated, V hi-only) since
// P in [0,1] is well-conditioned. Q tile staged via async global->LDS copy.
// ---------------------------------------------------------------------------

typedef __attribute__((ext_vector_type(16))) __bf16 v16bf;
typedef __attribute__((ext_vector_type(8)))  float  v8f;
typedef __attribute__((ext_vector_type(4)))  int    v4i;

#define Bn 4
#define Hn 16
#define Sn 2048
#define Dn 64

#if defined(__HIP_DEVICE_COMPILE__) && defined(__has_builtin)
#if __has_builtin(__builtin_amdgcn_global_load_async_to_lds_b128) && \
    __has_builtin(__builtin_amdgcn_s_wait_asynccnt)
#define HAVE_ASYNC_LDS 1
#endif
#endif
#ifndef HAVE_ASYNC_LDS
#define HAVE_ASYNC_LDS 0
#endif

static __device__ __forceinline__ v8f wmma_bf16(v16bf a, v16bf b, v8f c) {
    // (neg_a, A, neg_b, B, c_mod, C, reuse_a, reuse_b)
    return __builtin_amdgcn_wmma_f32_16x16x32_bf16(false, a, false, b,
                                                   (short)0, c, false, false);
}

static __device__ __forceinline__ void split_bf16(float f, __bf16& hi, __bf16& lo) {
    hi = (__bf16)f;
    lo = (__bf16)(f - (float)hi);
}

__global__ __launch_bounds__(256) void
attn_fused_kernel(const float* __restrict__ Q,
                  const float* __restrict__ K,
                  const float* __restrict__ V,
                  const unsigned char* __restrict__ Mask,
                  float* __restrict__ outR,   // [B,H,S,D]
                  float* __restrict__ outS)   // [B,H,S,S]
{
    // ---- LDS: score stripe + Q tile + output accumulator + reductions
    __shared__ float sS[16 * Sn];      // 128 KB
    __shared__ float sQ[16 * Dn];      // 4 KB (async-staged Q tile)
    __shared__ float sOut[16 * Dn];    // 4 KB
    __shared__ float sMax[16];
    __shared__ float sSum[16];

    const int tid   = threadIdx.x;
    const int wave  = tid >> 5;        // 0..7
    const int lane  = tid & 31;
    const int laneN = lane & 15;       // column within 16-wide tile (C layout)
    const int half  = lane >> 4;       // 0 or 1 (lane group)

    const int bid = blockIdx.x;        // 0..8191
    const int bh  = bid >> 7;          // (b*H + h), 0..63
    const int q0  = (bid & 127) << 4;  // first query row of this block
    const int b   = bh >> 4;

    const float* Qp = Q + (size_t)bh * Sn * Dn;
    const float* Kp = K + (size_t)bh * Sn * Dn;
    const float* Vp = V + (size_t)bh * Sn * Dn;
    const unsigned char* Mp = Mask + (size_t)b * Sn * Sn + (size_t)q0 * Sn;
    float* Rp = outR + (size_t)bh * Sn * Dn + (size_t)q0 * Dn;
    float* Sp = outS + (size_t)bh * Sn * Sn + (size_t)q0 * Sn;

    // ---- phase 0: zero output accumulator; async-stage Q tile into LDS.
    for (int i = tid; i < 16 * Dn; i += 256) sOut[i] = 0.0f;
#if HAVE_ASYNC_LDS
    {
        // 16 rows x 64 floats = 4 KB contiguous; 256 threads x 16 B each.
        typedef __attribute__((address_space(1))) v4i GV4;
        typedef __attribute__((address_space(3))) v4i LV4;
        GV4* g = (GV4*)(Qp + (size_t)q0 * Dn) + tid;
        LV4* l = (LV4*)sQ + tid;
        __builtin_amdgcn_global_load_async_to_lds_b128(g, l, 0, 0);
        __builtin_amdgcn_s_wait_asynccnt(0);
    }
#else
    {
        const float4* g = (const float4*)(Qp + (size_t)q0 * Dn);
        float4* l = (float4*)sQ;
        l[tid] = g[tid];
    }
#endif
    __syncthreads();

    // ---- build A-operand registers from the LDS Q tile (kept all kernel)
    // A layout (16-bit A 16x32): lane m = laneN; lanes 16-31 carry k+16 offset;
    // elems 0..7 -> k = base+0..7, elems 8..15 -> k = base+8..15, base=half*16
    v16bf aHi[2], aLo[2];
    {
        const float* qrow = &sQ[laneN * Dn];
        const int base = half * 16;
#pragma unroll
        for (int c = 0; c < 2; ++c) {
            const float* p0 = qrow + c * 32 + base;       // k = base..base+7
            const float* p1 = qrow + c * 32 + base + 8;   // k = base+8..base+15
#pragma unroll
            for (int i = 0; i < 8; ++i) {
                __bf16 h, l;
                split_bf16(p0[i], h, l); aHi[c][i]     = h; aLo[c][i]     = l;
                split_bf16(p1[i], h, l); aHi[c][8 + i] = h; aLo[c][8 + i] = l;
            }
        }
    }

    // ---- phase 1: w = (Q K^T)/sqrt(d) with mask, staged into LDS
    // wave owns 16 contiguous 16-col tiles: cols [wave*256, wave*256+255]
#pragma unroll 1
    for (int t = 0; t < 16; ++t) {
        const int kr0 = (wave * 16 + t) * 16;
        v8f acc = {0.f, 0.f, 0.f, 0.f, 0.f, 0.f, 0.f, 0.f};
#pragma unroll
        for (int c = 0; c < 2; ++c) {
            // B layout (32x16): lane n = laneN, k = half*16 + i  (contiguous d)
            const float* bp = Kp + (size_t)(kr0 + laneN) * Dn + c * 32 + half * 16;
            v16bf bHi, bLo;
#pragma unroll
            for (int i = 0; i < 16; ++i) {
                __bf16 h, l; split_bf16(bp[i], h, l);
                bHi[i] = h; bLo[i] = l;
            }
            acc = wmma_bf16(aHi[c], bHi, acc);   // hi*hi
            acc = wmma_bf16(aHi[c], bLo, acc);   // hi*lo
            acc = wmma_bf16(aLo[c], bHi, acc);   // lo*hi
        }
        if (t + 1 < 16) {
            // hint next K tile into cache (global_prefetch_b8)
            __builtin_prefetch(Kp + (size_t)((wave * 16 + t + 1) * 16 + laneN) * Dn, 0, 1);
        }
        // C layout: elem v holds row m = v + half*8, col = kr0 + laneN
#pragma unroll
        for (int v = 0; v < 8; ++v) {
            const int m = v + half * 8;
            float wv = acc[v] * 0.125f;          // 1/sqrt(64)
            if (Mp[(size_t)m * Sn + kr0 + laneN]) wv = -__builtin_inff();
            sS[m * Sn + kr0 + laneN] = wv;
        }
    }
    __syncthreads();

    // ---- phase 2: per-row max and sum(exp) over the 2048-wide stripe
    {
        const int row = tid >> 4;     // 16 threads per row
        const int i0  = tid & 15;
        float mx = -__builtin_inff();
        for (int c = i0; c < Sn; c += 16) mx = fmaxf(mx, sS[row * Sn + c]);
#pragma unroll
        for (int off = 8; off >= 1; off >>= 1)
            mx = fmaxf(mx, __shfl_xor(mx, off, 16));
        if (i0 == 0) sMax[row] = mx;
        __syncthreads();
        const float rm = sMax[row];
        float sum = 0.f;
        for (int c = i0; c < Sn; c += 16) sum += __expf(sS[row * Sn + c] - rm);
#pragma unroll
        for (int off = 8; off >= 1; off >>= 1)
            sum += __shfl_xor(sum, off, 16);
        if (i0 == 0) sSum[row] = sum;
    }
    __syncthreads();

    // ---- phase 3: normalize in place + stream probabilities to HBM (once)
    float rmax[8], rinv[8];
#pragma unroll
    for (int v = 0; v < 8; ++v) {
        const int m = v + half * 8;
        rmax[v] = sMax[m];
        rinv[v] = 1.0f / sSum[m];
    }
#pragma unroll 1
    for (int t = 0; t < 16; ++t) {
        const int col = (wave * 16 + t) * 16 + laneN;
#pragma unroll
        for (int v = 0; v < 8; ++v) {
            const int m = v + half * 8;
            const float p = __expf(sS[m * Sn + col] - rmax[v]) * rinv[v];
            sS[m * Sn + col] = p;
            Sp[(size_t)m * Sn + col] = p;
        }
    }
    __syncthreads();

    // ---- phase 4: partial P*V over this wave's 256 columns (K chunks of 32)
    // P is error-compensated (hi+lo); V uses hi only (P in [0,1] keeps the
    // product well-conditioned) -> 2 WMMAs per d-tile instead of 3.
    v8f accO[4];
#pragma unroll
    for (int dt = 0; dt < 4; ++dt)
        accO[dt] = (v8f){0.f, 0.f, 0.f, 0.f, 0.f, 0.f, 0.f, 0.f};

#pragma unroll 1
    for (int r = 0; r < 8; ++r) {
        const int col0 = wave * 256 + r * 32;
        // A = P chunk, same 16-bit A layout as phase 1
        v16bf pHi, pLo;
        {
            const float* p0 = &sS[laneN * Sn + col0 + half * 16];
            const float* p1 = p0 + 8;
#pragma unroll
            for (int i = 0; i < 8; ++i) {
                __bf16 h, l;
                split_bf16(p0[i], h, l); pHi[i]     = h; pLo[i]     = l;
                split_bf16(p1[i], h, l); pHi[8 + i] = h; pLo[8 + i] = l;
            }
        }
#pragma unroll
        for (int dt = 0; dt < 4; ++dt) {
            // B = V chunk (32x16): lane n = laneN, k = half*16 + i
            const float* vb = Vp + (size_t)(col0 + half * 16) * Dn + dt * 16 + laneN;
            v16bf vHi;
#pragma unroll
            for (int i = 0; i < 16; ++i)
                vHi[i] = (__bf16)vb[(size_t)i * Dn];
            accO[dt] = wmma_bf16(pHi, vHi, accO[dt]);
            accO[dt] = wmma_bf16(pLo, vHi, accO[dt]);
        }
    }
    // cross-wave reduction via LDS float atomics (ds_add_f32)
#pragma unroll
    for (int dt = 0; dt < 4; ++dt) {
#pragma unroll
        for (int v = 0; v < 8; ++v) {
            const int m = v + half * 8;
            atomicAdd(&sOut[m * Dn + dt * 16 + laneN], accO[dt][v]);
        }
    }
    __syncthreads();

    // ---- phase 5: write the 16x64 result block (contiguous row-major)
    for (int i = tid; i < 16 * Dn; i += 256) Rp[i] = sOut[i];
}

extern "C" void kernel_launch(void* const* d_in, const int* in_sizes, int n_in,
                              void* d_out, int out_size, void* d_ws, size_t ws_size,
                              hipStream_t stream) {
    (void)in_sizes; (void)n_in; (void)out_size; (void)d_ws; (void)ws_size;

    const float* q = (const float*)d_in[0];
    const float* k = (const float*)d_in[1];
    const float* v = (const float*)d_in[2];
    const unsigned char* m = (const unsigned char*)d_in[3];  // bool mask, 1B

    float* out_result = (float*)d_out;                              // [B,H,S,D]
    float* out_score  = out_result + (size_t)Bn * Hn * Sn * Dn;     // [B,H,S,S]

    const int grid = Bn * Hn * (Sn / 16);   // 8192 workgroups
    attn_fused_kernel<<<grid, 256, 0, stream>>>(q, k, v, m, out_result, out_score);
}
